// JANET_19799799235316
// MI455X (gfx1250) — matrix operand order repaired and checked by reading, not compiled, not run on hardware
//
#include <hip/hip_runtime.h>

#define TT  512
#define BB  64
#define II  512
#define HH  1024
#define GG  2048   // 2*H
#define NWG 32
#define PADW 1032  // 1024 + 8 bf16 pad -> LDS row stride 2064B, lane n hits bank (4n)%64

typedef __attribute__((ext_vector_type(16))) __bf16 bf16x16;
typedef __attribute__((ext_vector_type(8)))  __bf16 bf16x8;
typedef __attribute__((ext_vector_type(8)))  float  f32x8;

// ---------- fragment loaders (layouts per CDNA5 ISA 7.12.2, wave32) ----------

// A (16-bit, 16x32): lane l (m = l&15, hs = l>>4) holds
//   elems 0..7  = Arow[m][k0 + 8*hs      .. +8)
//   elems 8..15 = Arow[m][k0 + 16 + 8*hs .. +8)
__device__ __forceinline__ bf16x16 load_a_frag(const __bf16* __restrict__ row,
                                               int k0, int hs) {
  bf16x8 lo = *(const bf16x8*)(row + k0 + 8 * hs);
  bf16x8 hi = *(const bf16x8*)(row + k0 + 16 + 8 * hs);
  return __builtin_shufflevector(lo, hi, 0, 1, 2, 3, 4, 5, 6, 7,
                                 8, 9, 10, 11, 12, 13, 14, 15);
}

// B (16-bit, 32x16): lane l (n = l&15, half = l>>4) holds K values
// [k0+8*half,+8) and [k0+16+8*half,+8) of column n == row (gbase+n) of the row-major weight matrix.
__device__ __forceinline__ bf16x16 load_b_frag(const __bf16* __restrict__ wrow,
                                               int k0, int half) {
  bf16x8 lo = *(const bf16x8*)(wrow + k0 + 8 * half);
  bf16x8 hi = *(const bf16x8*)(wrow + k0 + 16 + 8 * half);
  return __builtin_shufflevector(lo, hi, 0, 1, 2, 3, 4, 5, 6, 7,
                                 8, 9, 10, 11, 12, 13, 14, 15);
}

// ---------- CDNA5 async global->LDS copy (ASYNCcnt-tracked, 16B/lane) -------
// INST_OFFSET is added to BOTH the LDS and global addresses (ISA 10.x async
// addressing), so one base address pair + immediate offsets covers the chunk.
template <int N>
struct AsyncCopy {
  static __device__ __forceinline__ void run(unsigned dst, const char* src) {
    AsyncCopy<N - 1>::run(dst, src);
    asm volatile("global_load_async_to_lds_b128 %0, %1, off offset:%2"
                 :: "v"(dst), "v"(src), "i"((N - 1) * 16)
                 : "memory");
  }
};
template <>
struct AsyncCopy<0> {
  static __device__ __forceinline__ void run(unsigned, const char*) {}
};

__device__ __forceinline__ void wait_asynccnt0() {
#if __has_builtin(__builtin_amdgcn_s_wait_asynccnt)
  __builtin_amdgcn_s_wait_asynccnt(0);
#else
  asm volatile("s_wait_asynccnt 0" ::: "memory");
#endif
}

// ---------- global (grid-wide) barrier for the persistent kernel ----------
__device__ __forceinline__ void grid_barrier(int* cnt, int* gen, int nwg) {
  __threadfence();
  __syncthreads();
  if (threadIdx.x == 0) {
    int g = __hip_atomic_load(gen, __ATOMIC_ACQUIRE, __HIP_MEMORY_SCOPE_AGENT);
    int prev = __hip_atomic_fetch_add(cnt, 1, __ATOMIC_ACQ_REL,
                                      __HIP_MEMORY_SCOPE_AGENT);
    if (prev == nwg - 1) {
      __hip_atomic_store(cnt, 0, __ATOMIC_RELAXED, __HIP_MEMORY_SCOPE_AGENT);
      __hip_atomic_store(gen, g + 1, __ATOMIC_RELEASE, __HIP_MEMORY_SCOPE_AGENT);
    } else {
      while (__hip_atomic_load(gen, __ATOMIC_ACQUIRE,
                               __HIP_MEMORY_SCOPE_AGENT) == g) {
        __builtin_amdgcn_s_sleep(2);
      }
    }
  }
  __syncthreads();
  __threadfence();
}

__device__ __forceinline__ float sigmoidf_fast(float x) {
  return 1.0f / (1.0f + __expf(-x));
}
__device__ __forceinline__ float tanhf_fast(float x) {
  return 1.0f - 2.0f / (__expf(2.0f * x) + 1.0f);
}

// ---------- prologue conversion kernels ----------
__global__ void f32_to_bf16_k(const float* __restrict__ s,
                              __bf16* __restrict__ d, int n) {
  for (int i = blockIdx.x * blockDim.x + threadIdx.x; i < n;
       i += gridDim.x * blockDim.x)
    d[i] = (__bf16)s[i];
}

__global__ void f32_copy_k(const float* __restrict__ s, float* __restrict__ d,
                           int n) {
  for (int i = blockIdx.x * blockDim.x + threadIdx.x; i < n;
       i += gridDim.x * blockDim.x)
    d[i] = s[i];
}

// ---------- persistent recurrent kernel ----------
// Grid: NWG=32 blocks x 256 threads (8 waves). Block wg owns h-columns
// [wg*32, wg*32+32) -> gate cols [wg*32,+32) (forget) and [H+wg*32,+32) (cand).
// LDS: w_hh slice (staged once) + h (staged per step, async, overlapped with
// the x-part GEMM). x / w_ih fragments stream from L2.
__global__ __launch_bounds__(256, 1) void janet_persistent(
    const __bf16* __restrict__ xbf,   // [T,B,I] bf16
    const float* __restrict__ b_ih, const float* __restrict__ b_hh,
    const __bf16* __restrict__ wih,   // [2H,I] bf16
    const __bf16* __restrict__ whh,   // [2H,H] bf16
    float* __restrict__ hf32,         // [B,H] f32 state
    __bf16* __restrict__ hb0, __bf16* __restrict__ hb1,  // double-buffered bf16 h
    float* __restrict__ out,          // [T,B,H] then [B,H] (h_last)
    int* __restrict__ bar) {
  extern __shared__ __align__(16) unsigned char smem_raw[];
  __bf16* whh_s = (__bf16*)smem_raw;            // [64][PADW]
  __bf16* h_s   = whh_s + 64 * PADW;            // [64][PADW]

  const int wg   = blockIdx.x;
  const int tid  = threadIdx.x;
  const int wave = tid >> 5;   // wave32
  const int lane = tid & 31;
  const int ln   = lane & 15;
  const int hs   = lane >> 4;

  const int mt   = wave >> 1;          // batch-row tile 0..3
  const int j    = wave & 1;           // h-col tile within block
  const int m0   = mt * 16;
  const int hcb  = wg * 32 + j * 16;   // base h-column of this wave's tiles
  const int arow = m0 + ln;            // this lane's activation row (batch idx)

  // --- stage the block's w_hh slice into LDS once (async copy) ---
  {
    const int r     = tid >> 2;
    const int chunk = tid & 3;
    const int grow  = (r < 32) ? (wg * 32 + r) : (HH + wg * 32 + (r - 32));
    const char* src = (const char*)(whh + (size_t)grow * HH + chunk * 256);
    unsigned dst = (unsigned)(uintptr_t)(whh_s + (size_t)r * PADW + chunk * 256);
    AsyncCopy<32>::run(dst, src);   // 512B per thread
    wait_asynccnt0();
  }
  __syncthreads();

  // B-fragment row pointers: w_hh from LDS, w_ih from global/L2
  const __bf16* wih_f  = wih + (size_t)(hcb + ln) * II;
  const __bf16* wih_c  = wih + (size_t)(HH + hcb + ln) * II;
  const __bf16* whhs_f = whh_s + (size_t)(j * 16 + ln) * PADW;
  const __bf16* whhs_c = whh_s + (size_t)(32 + j * 16 + ln) * PADW;
  const __bf16* hs_row = h_s + (size_t)arow * PADW;   // A rows for h part

  const float biasf = b_ih[hcb + ln] + b_hh[hcb + ln];
  const float biasc = b_ih[HH + hcb + ln] + b_hh[HH + hcb + ln];

  // h staging assignment: 512B per thread per step
  const int hr     = tid >> 2;
  const int hchunk = tid & 3;
  const unsigned hdst =
      (unsigned)(uintptr_t)(h_s + (size_t)hr * PADW + hchunk * 256);

  for (int t = 0; t < TT; ++t) {
    const __bf16* __restrict__ hin  = (t & 1) ? hb1 : hb0;
    __bf16* __restrict__       hout = (t & 1) ? hb0 : hb1;

    // --- issue async stage of h_{t-1} into LDS; completion waited AFTER the
    // x-part GEMM so the 128KB copy is hidden behind WMMA work. Safe vs. the
    // previous step's readers: the post-h-loop __syncthreads below ordered all
    // h_s reads before this point.
    AsyncCopy<32>::run(hdst,
                       (const char*)(hin + (size_t)hr * HH + hchunk * 256));

    const __bf16* __restrict__ xrow =
        xbf + (size_t)t * (BB * II) + (size_t)arow * II;
    // next step's x row for prefetch (clamped: never speculate past buffer)
    const __bf16* __restrict__ xrow_next =
        (t + 1 < TT) ? (xrow + (size_t)BB * II) : xrow;

    f32x8 accf = {0.f, 0.f, 0.f, 0.f, 0.f, 0.f, 0.f, 0.f};
    f32x8 accc = {0.f, 0.f, 0.f, 0.f, 0.f, 0.f, 0.f, 0.f};

    // K chunks 0..15: input projection (x_t @ w_ih^T), A+B from L2
#pragma unroll 2
    for (int k0 = 0; k0 < II; k0 += 32) {
      __builtin_prefetch(xrow_next + k0, 0, 3);   // global_prefetch_b8
      bf16x16 a  = load_a_frag(xrow, k0, hs);
      bf16x16 bf = load_b_frag(wih_f, k0, hs);
      bf16x16 bc = load_b_frag(wih_c, k0, hs);
      accf = __builtin_amdgcn_wmma_f32_16x16x32_bf16(false, a, false, bf,
                                                     (short)0, accf, false, false);
      accc = __builtin_amdgcn_wmma_f32_16x16x32_bf16(false, a, false, bc,
                                                     (short)0, accc, false, false);
    }

    // h copy must be complete (own waits + barrier for other waves' chunks)
    wait_asynccnt0();
    __syncthreads();

    // K chunks 16..47: recurrence (h @ w_hh^T), A+B from LDS
#pragma unroll 2
    for (int k0 = 0; k0 < HH; k0 += 32) {
      bf16x16 a  = load_a_frag(hs_row, k0, hs);
      bf16x16 bf = load_b_frag(whhs_f, k0, hs);
      bf16x16 bc = load_b_frag(whhs_c, k0, hs);
      accf = __builtin_amdgcn_wmma_f32_16x16x32_bf16(false, a, false, bf,
                                                     (short)0, accf, false, false);
      accc = __builtin_amdgcn_wmma_f32_16x16x32_bf16(false, a, false, bc,
                                                     (short)0, accc, false, false);
    }

    // all waves done reading h_s before next step's async copy overwrites it
    __syncthreads();

    // Pointwise epilogue. D layout: VGPR r -> row m0 + r + 8*hs, col = ln.
    float* __restrict__ outt = out + (size_t)t * (BB * HH);
    const int col = hcb + ln;
#pragma unroll
    for (int r = 0; r < 8; ++r) {
      const int m = m0 + r + 8 * hs;
      const size_t idx = (size_t)m * HH + col;
      const float fg   = sigmoidf_fast(accf[r] + biasf);
      const float cg   = tanhf_fast(accc[r] + biasc);
      const float hold = hf32[idx];
      const float hn   = fg * hold + (1.0f - fg) * cg;
      hf32[idx] = hn;
      hout[idx] = (__bf16)hn;
      outt[idx] = hn;
      if (t == TT - 1) out[(size_t)TT * BB * HH + idx] = hn;  // h_last
    }

    grid_barrier(bar, bar + 1, NWG);
  }
}

// ---------- host launcher ----------
extern "C" void kernel_launch(void* const* d_in, const int* in_sizes, int n_in,
                              void* d_out, int out_size, void* d_ws,
                              size_t ws_size, hipStream_t stream) {
  (void)in_sizes; (void)n_in; (void)out_size; (void)ws_size;
  const float* x    = (const float*)d_in[0];
  const float* h0   = (const float*)d_in[1];
  const float* w_ih = (const float*)d_in[2];
  const float* w_hh = (const float*)d_in[3];
  const float* b_ih = (const float*)d_in[4];
  const float* b_hh = (const float*)d_in[5];
  float* out = (float*)d_out;

  uint8_t* ws = (uint8_t*)d_ws;
  size_t off = 0;
  auto carve = [&](size_t bytes) -> uint8_t* {
    uint8_t* p = ws + off;
    off = (off + bytes + 255) & ~(size_t)255;
    return p;
  };
  int*    bar  = (int*)carve(8);
  float*  hf32 = (float*)carve((size_t)BB * HH * sizeof(float));
  __bf16* hb0  = (__bf16*)carve((size_t)BB * HH * sizeof(__bf16));
  __bf16* hb1  = (__bf16*)carve((size_t)BB * HH * sizeof(__bf16));
  __bf16* wihb = (__bf16*)carve((size_t)GG * II * sizeof(__bf16));
  __bf16* whhb = (__bf16*)carve((size_t)GG * HH * sizeof(__bf16));
  __bf16* xbf  = (__bf16*)carve((size_t)TT * BB * II * sizeof(__bf16));

  const int shbytes = 2 * 64 * PADW * (int)sizeof(__bf16);  // ~258 KB of 320 KB
  (void)hipFuncSetAttribute((const void*)janet_persistent,
                            hipFuncAttributeMaxDynamicSharedMemorySize, shbytes);

  hipMemsetAsync(bar, 0, 8, stream);  // reset barrier state every call

  f32_to_bf16_k<<<2048, 256, 0, stream>>>(x, xbf, TT * BB * II);
  f32_to_bf16_k<<<256, 256, 0, stream>>>(w_ih, wihb, GG * II);
  f32_to_bf16_k<<<256, 256, 0, stream>>>(w_hh, whhb, GG * HH);
  f32_to_bf16_k<<<64, 256, 0, stream>>>(h0, hb0, BB * HH);
  f32_to_bf16_k<<<64, 256, 0, stream>>>(h0, hb1, BB * HH);
  f32_copy_k<<<64, 256, 0, stream>>>(h0, hf32, BB * HH);

  janet_persistent<<<NWG, 256, shbytes, stream>>>(xbf, b_ih, b_hh, wihb, whhb,
                                                  hf32, hb0, hb1, out, bar);
}
